// BipartiteKernel_62019327754369
// MI455X (gfx1250) — matrix operand order repaired and checked by reading, not compiled, not run on hardware
//
#include <hip/hip_runtime.h>
#include <hip/hip_bf16.h>
#include <math.h>

typedef __attribute__((ext_vector_type(16))) _Float16 v16h;
typedef __attribute__((ext_vector_type(4)))  _Float16 v4h;
typedef __attribute__((ext_vector_type(8)))  float    v8f;

#define LATENT 128
#define HEADS  4

// float atomic max via integer atomics (handles negative values correctly)
__device__ __forceinline__ void atomicMaxF(float* addr, float v) {
    if (v >= 0.0f) {
        atomicMax((int*)addr, __float_as_int(v));
    } else {
        atomicMin((unsigned int*)addr, __float_as_uint(v));
    }
}

// ---------------------------------------------------------------- init
__global__ void k_init(float* __restrict__ out, float* __restrict__ qmax,
                       float* __restrict__ den, int nOut, int nQH) {
    int i = blockIdx.x * blockDim.x + threadIdx.x;
    int stride = gridDim.x * blockDim.x;
    for (int j = i; j < nOut; j += stride) out[j] = 0.0f;
    for (int j = i; j < nQH; j += stride) { qmax[j] = -__builtin_inff(); den[j] = 0.0f; }
}

// ---------------------------------------------------------------- pack weights into wave32 B-fragment layout
// B fragment (f16, 32x16): lane L holds column n = nt*16 + (L&15); its 16 halves are
// K = (L>=16 ? 8 : 0) + (i&7) + (i>=8 ? 16 : 0)  within the 32-wide K chunk.
__global__ void k_pack(const float* __restrict__ Wv, const float* __restrict__ W1,
                       _Float16* __restrict__ WvB, _Float16* __restrict__ W1B) {
    int idx = blockIdx.x * blockDim.x + threadIdx.x;
    if (idx < 4 * 8 * 32 * 16) {                      // Wv: 4 kchunks x 8 ntiles
        int i = idx & 15, lane = (idx >> 4) & 31, nt = (idx >> 9) & 7, kc = idx >> 12;
        int n = nt * 16 + (lane & 15);
        int k = kc * 32 + ((lane >> 4) * 8) + (i & 7) + ((i >> 3) * 16);
        WvB[idx] = (_Float16)Wv[k * LATENT + n];
    }
    int idx2 = idx - 4 * 8 * 32 * 16;
    if (idx2 >= 0 && idx2 < 8 * 32 * 16) {            // W1: 1 kchunk (K=9 padded to 32) x 8 ntiles
        int i = idx2 & 15, lane = (idx2 >> 4) & 31, nt = (idx2 >> 9) & 7;
        int n = nt * 16 + (lane & 15);
        int k = ((lane >> 4) * 8) + (i & 7) + ((i >> 3) * 16);
        W1B[idx2] = (k < 9) ? (_Float16)W1[k * LATENT + n] : (_Float16)0.0f;
    }
}

// ---------------------------------------------------------------- V = h_obs @ Wv + bv  (WMMA f16, f16 output)
__global__ void k_value(const float* __restrict__ h_obs, const _Float16* __restrict__ WvB,
                        const float* __restrict__ bv, _Float16* __restrict__ V,
                        int nO, int mtiles) {
    int wid  = (blockIdx.x * blockDim.x + threadIdx.x) >> 5;
    int lane = threadIdx.x & 31;
    if (wid >= mtiles) return;                       // uniform per wave: EXEC stays all-ones
    int l = lane & 15, g = lane >> 4;

    v8f c[8];
#pragma unroll
    for (int nt = 0; nt < 8; ++nt)
#pragma unroll
        for (int r = 0; r < 8; ++r) c[nt][r] = 0.0f;   // first WMMA folds to inline-0 SRC2

    int mrow = wid * 16 + l;
    if (mrow > nO - 1) mrow = nO - 1;                // clamp (keeps EXEC full)
    const float* rowp = h_obs + (size_t)mrow * LATENT;

#pragma unroll
    for (int kc = 0; kc < 4; ++kc) {
        // A fragment (f16, 16x32): lane half g shifts K by 8; halves i>=8 map to K+16.
        const float4* p = (const float4*)(rowp + kc * 32 + g * 8);
        float4 x0 = p[0], x1 = p[1], x2 = p[4], x3 = p[5];
        v16h a;
        a[0]=(_Float16)x0.x; a[1]=(_Float16)x0.y; a[2]=(_Float16)x0.z; a[3]=(_Float16)x0.w;
        a[4]=(_Float16)x1.x; a[5]=(_Float16)x1.y; a[6]=(_Float16)x1.z; a[7]=(_Float16)x1.w;
        a[8]=(_Float16)x2.x; a[9]=(_Float16)x2.y; a[10]=(_Float16)x2.z; a[11]=(_Float16)x2.w;
        a[12]=(_Float16)x3.x; a[13]=(_Float16)x3.y; a[14]=(_Float16)x3.z; a[15]=(_Float16)x3.w;
        // preload all 8 B fragments; barrier keeps them live so WMMAs issue back-to-back
        v16h b[8];
#pragma unroll
        for (int nt = 0; nt < 8; ++nt)
            b[nt] = *(const v16h*)(WvB + ((size_t)((kc * 8 + nt) * 32 + lane)) * 16);
        __builtin_amdgcn_sched_barrier(0);
#pragma unroll
        for (int nt = 0; nt < 8; ++nt)
            c[nt] = __builtin_amdgcn_wmma_f32_16x16x32_f16(false, a, false, b[nt],
                                                           (short)0, c[nt], false, false);
    }

    float bvn[8];
#pragma unroll
    for (int nt = 0; nt < 8; ++nt) bvn[nt] = bv[nt * 16 + l];
#pragma unroll
    for (int r = 0; r < 8; ++r) {
        int m = wid * 16 + r + 8 * g;                // C/D layout: M = r + 8*(lane>=16)
        if (m < nO) {
            _Float16* vrow = V + (size_t)m * LATENT + l;
#pragma unroll
            for (int nt = 0; nt < 8; ++nt) vrow[nt * 16] = (_Float16)(c[nt][r] + bvn[nt]);
        }
    }
}

// ---------------------------------------------------------------- per-edge MLP + logits (WMMA f16)
__global__ void k_edge(const float* __restrict__ pos_obs, const float* __restrict__ pos_query,
                       const int* __restrict__ src, const int* __restrict__ dst,
                       const _Float16* __restrict__ W1B, const float* __restrict__ b1,
                       const float* __restrict__ W2, const float* __restrict__ b2,
                       const float* __restrict__ logsig,
                       float* __restrict__ lg, float* __restrict__ qmax,
                       int E, int tiles) {
    int wid  = (blockIdx.x * blockDim.x + threadIdx.x) >> 5;
    int lane = threadIdx.x & 31;
    if (wid >= tiles) return;                        // uniform per wave
    int l = lane & 15, g = lane >> 4;

    int e = wid * 16 + l;
    if (e > E - 1) e = E - 1;                        // clamp for partial tiles
    int s = src[e], d = dst[e];
    float ox = pos_obs[s * 3 + 0], oy = pos_obs[s * 3 + 1], oz = pos_obs[s * 3 + 2];
    float qx = pos_query[d * 3 + 0], qy = pos_query[d * 3 + 1], qz = pos_query[d * 3 + 2];
    float rx = qx - ox, ry = qy - oy, rz = qz - oz;
    float dist2 = rx * rx + ry * ry + rz * rz;

    // A fragment 16x32: lanes 0-15 hold K=0..7 (+16..23=0); lanes 16-31 hold K=8 (rest 0).
    v16h a;
#pragma unroll
    for (int i = 0; i < 16; ++i) a[i] = (_Float16)0.0f;
    if (g == 0) {
        a[0]=(_Float16)rx; a[1]=(_Float16)ry; a[2]=(_Float16)rz;
        a[3]=(_Float16)qx; a[4]=(_Float16)qy; a[5]=(_Float16)qz;
        a[6]=(_Float16)ox; a[7]=(_Float16)oy;
    } else {
        a[0]=(_Float16)oz;
    }

    // hid = relu(edge_attr @ W1 + b1), kept entirely in C fragments (C starts at inline 0)
    v16h b[8];
#pragma unroll
    for (int nt = 0; nt < 8; ++nt)
        b[nt] = *(const v16h*)(W1B + (size_t)(nt * 32 + lane) * 16);
    __builtin_amdgcn_sched_barrier(0);

    v8f cf[8];
#pragma unroll
    for (int nt = 0; nt < 8; ++nt) {
        v8f c;
#pragma unroll
        for (int r = 0; r < 8; ++r) c[r] = 0.0f;
        cf[nt] = __builtin_amdgcn_wmma_f32_16x16x32_f16(false, a, false, b[nt],
                                                        (short)0, c, false, false);
    }
    // bias + relu in epilogue
#pragma unroll
    for (int nt = 0; nt < 8; ++nt) {
        float b1n = b1[nt * 16 + l];
#pragma unroll
        for (int r = 0; r < 8; ++r) cf[nt][r] = fmaxf(cf[nt][r] + b1n, 0.0f);
    }

    // logits = hid @ W2 : K is striped across the 16-lane group in C layout,
    // so do per-lane FMAs then a 16-lane butterfly reduction.
    const float4* W2v = (const float4*)W2;
    float acc[8][4];
#pragma unroll
    for (int r = 0; r < 8; ++r) { acc[r][0]=0.f; acc[r][1]=0.f; acc[r][2]=0.f; acc[r][3]=0.f; }
#pragma unroll
    for (int t = 0; t < 8; ++t) {
        float4 w = W2v[t * 16 + l];
#pragma unroll
        for (int r = 0; r < 8; ++r) {
            float hv = cf[t][r];
            acc[r][0] += hv * w.x; acc[r][1] += hv * w.y;
            acc[r][2] += hv * w.z; acc[r][3] += hv * w.w;
        }
    }
#pragma unroll
    for (int r = 0; r < 8; ++r)
#pragma unroll
        for (int h = 0; h < 4; ++h) {
            float v = acc[r][h];
            v += __shfl_xor(v, 1, 32); v += __shfl_xor(v, 2, 32);
            v += __shfl_xor(v, 4, 32); v += __shfl_xor(v, 8, 32);
            acc[r][h] = v;
        }

    float d2s[8]; int dsts[8];
#pragma unroll
    for (int r = 0; r < 8; ++r) {
        int m = r + 8 * g;
        d2s[r]  = __shfl(dist2, m, 32);              // lane m loaded edge row m
        dsts[r] = __shfl(d, m, 32);
    }

    float ls = logsig[0];
    float sg = expf(ls) + 1e-6f;
    float inv2s2 = 1.0f / (2.0f * sg * sg);
    float4 b2v = *(const float4*)b2;

    if (l < 4) {                                     // lanes l=0..3 write head l for rows m=r+8g
        float b2l = (l == 0) ? b2v.x : (l == 1) ? b2v.y : (l == 2) ? b2v.z : b2v.w;
#pragma unroll
        for (int r = 0; r < 8; ++r) {
            int ee = wid * 16 + r + 8 * g;
            if (ee < E) {
                float av = (l == 0) ? acc[r][0] : (l == 1) ? acc[r][1]
                         : (l == 2) ? acc[r][2] : acc[r][3];
                float val = av + b2l - d2s[r] * inv2s2;
                lg[ee * 4 + l] = val;
                atomicMaxF(&qmax[dsts[r] * 4 + l], val);
            }
        }
    }
}

// ---------------------------------------------------------------- ex = exp(logit - max); denom += ex
// dst is sorted: 4-lane groups sweep 32-edge sub-ranges, run-length flush to cut atomics.
__global__ void k_softmax(const int* __restrict__ dst, float* __restrict__ lg,
                          const float* __restrict__ qmax, float* __restrict__ den,
                          int E, int nwaves, int epw) {
    int wid  = (blockIdx.x * blockDim.x + threadIdx.x) >> 5;
    int lane = threadIdx.x & 31;
    if (wid >= nwaves) return;
    int grp = lane >> 2, h = lane & 3;               // 8 groups of 4 lanes; lane owns head h
    int sub = epw >> 3;                              // edges per group
    int s0 = wid * epw + grp * sub;
    int s1 = s0 + sub; if (s1 > E) s1 = E;

    float acc = 0.0f; int cur = -1; float mx = 0.0f;
    for (int e = s0; e < s1; ++e) {
        int d = dst[e];
        if (d != cur) {
            if (cur >= 0) atomicAdd(&den[cur * 4 + h], acc);
            acc = 0.0f; cur = d;
            mx = qmax[d * 4 + h];
        }
        float ex = __expf(lg[e * 4 + h] - mx);
        lg[e * 4 + h] = ex;                          // overwrite logits with ex
        acc += ex;
    }
    if (cur >= 0) atomicAdd(&den[cur * 4 + h], acc);
}

// ---------------------------------------------------------------- out[dst] += attn * V[src]  (dst sorted -> run-length flush)
__global__ void k_scatter(const int* __restrict__ src, const int* __restrict__ dst,
                          const float* __restrict__ ex, const float* __restrict__ den,
                          const _Float16* __restrict__ V, float* __restrict__ out,
                          int E, int nwaves, int epw) {
    int wid  = (blockIdx.x * blockDim.x + threadIdx.x) >> 5;
    int lane = threadIdx.x & 31;
    if (wid >= nwaves) return;
    int e0 = wid * epw;
    int e1 = e0 + epw; if (e1 > E) e1 = E;
    int h = lane >> 3;                               // lane owns dims [lane*4, lane*4+4), head = dim/32

    float ax = 0.f, ay = 0.f, az = 0.f, aw = 0.f;
    int cur = -1; float dinv = 0.f;
    for (int e = e0; e < e1; ++e) {
        int d = dst[e];                              // uniform across wave
        if (d != cur) {
            if (cur >= 0) {
                float* o = out + (size_t)cur * LATENT + lane * 4;
                atomicAdd(o + 0, ax); atomicAdd(o + 1, ay);
                atomicAdd(o + 2, az); atomicAdd(o + 3, aw);
            }
            ax = ay = az = aw = 0.f; cur = d;
            dinv = 1.0f / (den[d * 4 + h] + 1e-16f);
        }
        float att = ex[e * 4 + h] * dinv;
        v4h v = *(const v4h*)(V + (size_t)src[e] * LATENT + lane * 4);   // 8B f16 load
        ax += att * (float)v[0]; ay += att * (float)v[1];
        az += att * (float)v[2]; aw += att * (float)v[3];
    }
    if (cur >= 0) {
        float* o = out + (size_t)cur * LATENT + lane * 4;
        atomicAdd(o + 0, ax); atomicAdd(o + 1, ay);
        atomicAdd(o + 2, az); atomicAdd(o + 3, aw);
    }
}

// ---------------------------------------------------------------- launch
extern "C" void kernel_launch(void* const* d_in, const int* in_sizes, int n_in,
                              void* d_out, int out_size, void* d_ws, size_t ws_size,
                              hipStream_t stream) {
    const float* h_obs     = (const float*)d_in[0];
    const float* pos_obs   = (const float*)d_in[1];
    const float* pos_query = (const float*)d_in[2];
    const int*   src       = (const int*)d_in[3];
    const int*   dst       = (const int*)d_in[4];
    const float* W1        = (const float*)d_in[5];
    const float* b1        = (const float*)d_in[6];
    const float* W2        = (const float*)d_in[7];
    const float* b2        = (const float*)d_in[8];
    const float* Wv        = (const float*)d_in[9];
    const float* bv        = (const float*)d_in[10];
    const float* logsig    = (const float*)d_in[11];

    int nO = in_sizes[0] / LATENT;
    int nQ = in_sizes[2] / 3;
    int E  = in_sizes[3];

    float* out = (float*)d_out;
    float* wsf = (float*)d_ws;

    // workspace layout (float units): V(f16) | logits/ex(f32) | qmax | denom | WvB(f16) | W1B(f16)
    size_t offV   = 0;
    size_t offLG  = offV  + ((size_t)nO * LATENT) / 2;   // V stored as f16
    size_t offQM  = offLG + (size_t)E * HEADS;
    size_t offDN  = offQM + (size_t)nQ * HEADS;
    size_t offWvB = (offDN + (size_t)nQ * HEADS + 15) & ~(size_t)15;   // 64B align
    size_t offW1B = offWvB + (size_t)(4 * 8 * 32 * 16) / 2;

    _Float16* Vbuf = (_Float16*)(wsf + offV);
    float*    LG   = wsf + offLG;
    float*    QM   = wsf + offQM;
    float*    DN   = wsf + offDN;
    _Float16* WvB  = (_Float16*)(wsf + offWvB);
    _Float16* W1B  = (_Float16*)(wsf + offW1B);

    k_init<<<4096, 256, 0, stream>>>(out, QM, DN, nQ * LATENT, nQ * HEADS);

    k_pack<<<(20480 + 255) / 256, 256, 0, stream>>>(Wv, W1, WvB, W1B);

    {
        int mtiles = (nO + 15) / 16;
        k_value<<<(mtiles + 7) / 8, 256, 0, stream>>>(h_obs, WvB, bv, Vbuf, nO, mtiles);
    }
    {
        int tiles = (E + 15) / 16;
        k_edge<<<(tiles + 7) / 8, 256, 0, stream>>>(pos_obs, pos_query, src, dst,
                                                    W1B, b1, W2, b2, logsig, LG, QM, E, tiles);
    }
    {
        int epw = 256;                               // 8 groups x 32 edges
        int nw = (E + epw - 1) / epw;
        k_softmax<<<(nw + 7) / 8, 256, 0, stream>>>(dst, LG, QM, DN, E, nw, epw);
    }
    {
        int epw = 128;
        int nw = (E + epw - 1) / epw;
        k_scatter<<<(nw + 7) / 8, 256, 0, stream>>>(src, dst, LG, DN, Vbuf, out, E, nw, epw);
    }
}
